// SeqModel_51977694216669
// MI455X (gfx1250) — matrix operand order repaired and checked by reading, not compiled, not run on hardware
//
#include <hip/hip_runtime.h>

// Model constants (from the reference)
#define DD   3        // model dim
#define DP   4        // padded model dim in LDS
#define FF   2048     // FFN hidden
#define NL   12       // layers
#define NB   16       // batch
#define NT   32       // seq len
#define NTOK 512      // NT*NB tokens
#define EPS  1e-5f

#define L2_ELEMS_PER_SIDE (NL * DD * FF)          // 73728
#define L2_TOTAL_ELEMS    (2 * L2_ELEMS_PER_SIDE) // 147456
#define WS_NEEDED_BYTES   (L2_TOTAL_ELEMS * 2)    // bf16

typedef __attribute__((ext_vector_type(16))) __bf16 v16bf;
typedef __attribute__((ext_vector_type(8)))  float  v8f;
typedef __attribute__((ext_vector_type(2)))  float  v2f;

struct P {
    const float *src, *angle;
    const float *esaw, *esab, *esaow, *esaob;
    const float *el1w, *el1b, *el2w, *el2b;
    const float *en1w, *en1b, *en2w, *en2b, *enfw, *enfb;
    const float *dsaw, *dsab, *dsaow, *dsaob;
    const float *dcaw, *dcab, *dcaow, *dcaob;
    const float *dl1w, *dl1b, *dl2w, *dl2b;
    const float *dn1w, *dn1b, *dn2w, *dn2b, *dn3w, *dn3b, *dnfw, *dnfb;
    const __bf16 *l2bf;   // packed bf16 l2 weights in workspace (enc then dec)
    float *out;
};

// ---- pre-pass: convert both l2_w tensors to bf16 once (reused 31x by decoder)
__global__ void pack_l2_bf16_kernel(const float* el2w, const float* dl2w, __bf16* dst) {
    for (int idx = blockIdx.x * blockDim.x + threadIdx.x;
         idx < L2_TOTAL_ELEMS; idx += gridDim.x * blockDim.x) {
        float v = (idx < L2_ELEMS_PER_SIDE) ? el2w[idx]
                                            : dl2w[idx - L2_ELEMS_PER_SIDE];
        dst[idx] = (__bf16)v;
    }
}

// LayerNorm over 3 elements, writes padded row (4th lane = 0)
__device__ __forceinline__ void ln3_store(float y0, float y1, float y2,
                                          const float* w, const float* b,
                                          float* dst4) {
    float m  = (y0 + y1 + y2) * (1.0f / 3.0f);
    float d0 = y0 - m, d1 = y1 - m, d2 = y2 - m;
    float v  = (d0 * d0 + d1 * d1 + d2 * d2) * (1.0f / 3.0f);
    float inv = rsqrtf(v + EPS);
    dst4[0] = d0 * inv * w[0] + b[0];
    dst4[1] = d1 * inv * w[1] + b[1];
    dst4[2] = d2 * inv * w[2] + b[2];
    dst4[3] = 0.0f;
}

// MHA (head_dim == 1 -> per-head scalar attention) + residual + LN, in place on xbuf.
__device__ __forceinline__ void run_attention(float* xbuf, const float* kvsrc,
                                              float* sk, float* sv,
                                              const float* w, const float* bias,
                                              const float* ow, const float* ob,
                                              const float* lnw, const float* lnb) {
    const int n    = threadIdx.x;      // token id = t*16 + b
    const int bcol = n & 15;           // batch index
    float xv0 = kvsrc[n * DP + 0], xv1 = kvsrc[n * DP + 1], xv2 = kvsrc[n * DP + 2];
    #pragma unroll
    for (int d = 0; d < 3; ++d) {
        sk[n * DP + d] = w[(3 + d) * 3 + 0] * xv0 + w[(3 + d) * 3 + 1] * xv1 +
                         w[(3 + d) * 3 + 2] * xv2 + bias[3 + d];
        sv[n * DP + d] = w[(6 + d) * 3 + 0] * xv0 + w[(6 + d) * 3 + 1] * xv1 +
                         w[(6 + d) * 3 + 2] * xv2 + bias[6 + d];
    }
    float xq0 = xbuf[n * DP + 0], xq1 = xbuf[n * DP + 1], xq2 = xbuf[n * DP + 2];
    float q[3];
    #pragma unroll
    for (int d = 0; d < 3; ++d)
        q[d] = w[d * 3 + 0] * xq0 + w[d * 3 + 1] * xq1 + w[d * 3 + 2] * xq2 + bias[d];
    __syncthreads();
    float o[3];
    #pragma unroll
    for (int h = 0; h < 3; ++h) {
        float mx = -1e30f;
        #pragma unroll
        for (int kp = 0; kp < NT; ++kp)
            mx = fmaxf(mx, q[h] * sk[(kp * NB + bcol) * DP + h]);
        float den = 0.0f, acc = 0.0f;
        #pragma unroll
        for (int kp = 0; kp < NT; ++kp) {
            int idx = (kp * NB + bcol) * DP + h;
            float e = __expf(q[h] * sk[idx] - mx);
            den += e;
            acc += e * sv[idx];
        }
        o[h] = acc / den;
    }
    float y[3];
    #pragma unroll
    for (int d = 0; d < 3; ++d)
        y[d] = xbuf[n * DP + d] + ob[d] +
               ow[d * 3 + 0] * o[0] + ow[d * 3 + 1] * o[1] + ow[d * 3 + 2] * o[2];
    ln3_store(y[0], y[1], y[2], lnw, lnb, &xbuf[n * DP]);
    __syncthreads();
}

// Fused FFN via WMMA:
//  GEMM1: x(512x3) @ l1w.T -> v_wmma_f32_16x16x4_f32 (exact f32, K=3 padded to 4)
//  GEMM2: relu(h)(512x2048) @ l2w.T -> v_wmma_f32_16x16x32_bf16, K=2048 contraction
template <bool PACKED>
__device__ __forceinline__ void run_ffn(float* xbuf, __bf16* stage, float* ffnout,
                                        const float* l1w, const float* l1b,
                                        const float* l2w, const __bf16* l2bf,
                                        const float* l2b,
                                        const float* lnw, const float* lnb) {
    const int tid  = threadIdx.x;
    const int wave = tid >> 5;
    const int lane = tid & 31;
    const int lm   = lane & 15;
    const int hb   = lane >> 4;
    __bf16* mystage = stage + wave * (16 * 32);

    __builtin_prefetch(l1w, 0, 0);   // pull layer weights toward the WGP early

    #pragma unroll 1
    for (int half_mt = 0; half_mt < 2; ++half_mt) {
        const int mt  = wave + half_mt * 16;
        const int row = mt * 16 + lm;

        // A1 fragment, f32 16x4 layout: lane m = lane&15;
        // VGPR0 holds K = (lane>>4)*2, VGPR1 holds K = (lane>>4)*2 + 1. Col 3 is 0-pad.
        v2f a1;
        a1.x = xbuf[row * DP + (hb ? 2 : 0)];
        a1.y = xbuf[row * DP + (hb ? 3 : 1)];   // col 3 == 0

        v8f acc2 = {0.f, 0.f, 0.f, 0.f, 0.f, 0.f, 0.f, 0.f};

        #pragma unroll 1
        for (int jj = 0; jj < FF / 32; ++jj) {    // 32 FFN columns per iteration
            #pragma unroll
            for (int hf = 0; hf < 2; ++hf) {
                const int j = jj * 2 + hf;        // 16-column tile
                // B1 fragment (4x16 f32): B[k][n] = l1w[n][k]; lane n = lane&15.
                const float* wp = l1w + (j * 16 + lm) * 3;
                v2f b1;
                b1.x = hb ? wp[2] : wp[0];
                b1.y = hb ? 0.0f  : wp[1];        // k=3 pad
                v8f c1 = {0.f, 0.f, 0.f, 0.f, 0.f, 0.f, 0.f, 0.f};
                c1 = __builtin_amdgcn_wmma_f32_16x16x4_f32(
                        false, a1, false, b1, (short)0, c1, false, false);
                const float bias = l1b[j * 16 + lm];
                // D-layout: vgpr r -> row r (lanes 0-15) / r+8 (lanes 16-31), col = lane&15
                #pragma unroll
                for (int r = 0; r < 8; ++r) {
                    float hv = fmaxf(c1[r] + bias, 0.0f);
                    mystage[(r + hb * 8) * 32 + hf * 16 + lm] = (__bf16)hv;
                }
            }
            // A2 fragment (h tile, K=32) gathered from staged rows in bf16 A-layout
            v16bf a2;
            #pragma unroll
            for (int v = 0; v < 8; ++v) {
                const int k = hb * 16 + v * 2;
                a2[2 * v + 0] = mystage[lm * 32 + k + 0];
                a2[2 * v + 1] = mystage[lm * 32 + k + 1];
            }
            // B2 fragment: B[k][d] = l2w[d][jj*32 + k]; lane d = lane&15, only d<3 nonzero
            v16bf b2;
            if (PACKED) {
                if (lm < 3) {
                    // 16 contiguous bf16 (32B, aligned): one vector load, no conversion
                    b2 = *reinterpret_cast<const v16bf*>(l2bf + lm * FF + jj * 32 + hb * 16);
                } else {
                    #pragma unroll
                    for (int e = 0; e < 16; ++e) b2[e] = (__bf16)0.0f;
                }
            } else {
                #pragma unroll
                for (int e = 0; e < 16; ++e) b2[e] = (__bf16)0.0f;
                if (lm < 3) {
                    const float* wp2 = l2w + lm * FF + jj * 32 + hb * 16;
                    #pragma unroll
                    for (int v = 0; v < 8; ++v) {
                        b2[2 * v + 0] = (__bf16)wp2[2 * v + 0];
                        b2[2 * v + 1] = (__bf16)wp2[2 * v + 1];
                    }
                }
            }
            acc2 = __builtin_amdgcn_wmma_f32_16x16x32_bf16(
                      false, a2, false, b2, (short)0, acc2, false, false);
        }

        if (lm < 3) {
            const float bias2 = l2b[lm];
            #pragma unroll
            for (int r = 0; r < 8; ++r)
                ffnout[(mt * 16 + r + hb * 8) * DP + lm] = acc2[r] + bias2;
        }
    }
    __syncthreads();
    const int n = tid;
    float y0 = xbuf[n * DP + 0] + ffnout[n * DP + 0];
    float y1 = xbuf[n * DP + 1] + ffnout[n * DP + 1];
    float y2 = xbuf[n * DP + 2] + ffnout[n * DP + 2];
    ln3_store(y0, y1, y2, lnw, lnb, &xbuf[n * DP]);
    __syncthreads();
}

template <bool PACKED>
__global__ __launch_bounds__(512, 1)
void seq2seq_persistent_kernel(P p) {
    __shared__ float xbuf[NTOK * DP];     // 8 KB   current activations
    __shared__ float membuf[NTOK * DP];   // 8 KB   encoder memory
    __shared__ float outbuf[NTOK * DP];   // 8 KB   decoder running output
    __shared__ __align__(32) unsigned char scratch[24 * 1024];
    float*  sk     = (float*)scratch;                       // [0,8K)   attn K
    float*  sv     = (float*)(scratch + 8 * 1024);          // [8K,16K) attn V
    __bf16* stage  = (__bf16*)scratch;                      // [0,16K)  FFN h staging
    float*  ffnout = (float*)(scratch + 16 * 1024);         // [16K,24K)

    const int n = threadIdx.x;            // one thread per token
    const int t = n >> 4;
    const int b = n & 15;

    // Build s (T,B,D): channels of src plus broadcast angle; out0 = s row 0.
    float s0 = p.src[b * 64 + 0 * 32 + t];
    float s1 = p.src[b * 64 + 1 * 32 + t];
    float s2 = p.angle[b];
    xbuf[n * DP + 0] = s0; xbuf[n * DP + 1] = s1;
    xbuf[n * DP + 2] = s2; xbuf[n * DP + 3] = 0.0f;
    outbuf[n * DP + 0] = (t == 0) ? s0 : 0.0f;
    outbuf[n * DP + 1] = (t == 0) ? s1 : 0.0f;
    outbuf[n * DP + 2] = (t == 0) ? s2 : 0.0f;
    outbuf[n * DP + 3] = 0.0f;
    __syncthreads();

    // ---------------- Encoder ----------------
    #pragma unroll 1
    for (int i = 0; i < NL; ++i) {
        run_attention(xbuf, xbuf, sk, sv,
                      p.esaw + i * 27, p.esab + i * 9, p.esaow + i * 9, p.esaob + i * 3,
                      p.en1w + i * 3, p.en1b + i * 3);
        run_ffn<PACKED>(xbuf, stage, ffnout,
                p.el1w + i * FF * 3, p.el1b + i * FF,
                p.el2w + i * 3 * FF, p.l2bf + i * 3 * FF, p.el2b + i * 3,
                p.en2w + i * 3, p.en2b + i * 3);
    }
    ln3_store(xbuf[n * DP + 0], xbuf[n * DP + 1], xbuf[n * DP + 2],
              p.enfw, p.enfb, &membuf[n * DP]);
    __syncthreads();

    // ---------------- Decoder: serial scan over 31 steps ----------------
    #pragma unroll 1
    for (int step = 1; step < NT; ++step) {
        #pragma unroll
        for (int d = 0; d < 4; ++d) xbuf[n * DP + d] = outbuf[n * DP + d];
        __syncthreads();
        #pragma unroll 1
        for (int i = 0; i < NL; ++i) {
            run_attention(xbuf, xbuf, sk, sv,          // self-attention
                          p.dsaw + i * 27, p.dsab + i * 9, p.dsaow + i * 9, p.dsaob + i * 3,
                          p.dn1w + i * 3, p.dn1b + i * 3);
            run_attention(xbuf, membuf, sk, sv,        // cross-attention
                          p.dcaw + i * 27, p.dcab + i * 9, p.dcaow + i * 9, p.dcaob + i * 3,
                          p.dn2w + i * 3, p.dn2b + i * 3);
            run_ffn<PACKED>(xbuf, stage, ffnout,
                    p.dl1w + i * FF * 3, p.dl1b + i * FF,
                    p.dl2w + i * 3 * FF,
                    p.l2bf + L2_ELEMS_PER_SIDE + i * 3 * FF, p.dl2b + i * 3,
                    p.dn3w + i * 3, p.dn3b + i * 3);
        }
        if (t == step)   // commit only pred[step] into the running output
            ln3_store(xbuf[n * DP + 0], xbuf[n * DP + 1], xbuf[n * DP + 2],
                      p.dnfw, p.dnfb, &outbuf[n * DP]);
        __syncthreads();
    }

    // Output: (B, 3, T)
    p.out[b * 96 + 0 * 32 + t] = outbuf[n * DP + 0];
    p.out[b * 96 + 1 * 32 + t] = outbuf[n * DP + 1];
    p.out[b * 96 + 2 * 32 + t] = outbuf[n * DP + 2];
}

extern "C" void kernel_launch(void* const* d_in, const int* in_sizes, int n_in,
                              void* d_out, int out_size, void* d_ws, size_t ws_size,
                              hipStream_t stream) {
    (void)in_sizes; (void)n_in; (void)out_size;
    P p;
    int k = 0;
    const float** slots[] = {
        &p.src, &p.angle,
        &p.esaw, &p.esab, &p.esaow, &p.esaob,
        &p.el1w, &p.el1b, &p.el2w, &p.el2b,
        &p.en1w, &p.en1b, &p.en2w, &p.en2b, &p.enfw, &p.enfb,
        &p.dsaw, &p.dsab, &p.dsaow, &p.dsaob,
        &p.dcaw, &p.dcab, &p.dcaow, &p.dcaob,
        &p.dl1w, &p.dl1b, &p.dl2w, &p.dl2b,
        &p.dn1w, &p.dn1b, &p.dn2w, &p.dn2b, &p.dn3w, &p.dn3b, &p.dnfw, &p.dnfb
    };
    for (auto s : slots) *s = (const float*)d_in[k++];
    p.out  = (float*)d_out;
    p.l2bf = (const __bf16*)d_ws;

    const bool packed = (ws_size >= (size_t)WS_NEEDED_BYTES) && (d_ws != nullptr);
    if (packed) {
        pack_l2_bf16_kernel<<<dim3(144), dim3(256), 0, stream>>>(p.el2w, p.dl2w,
                                                                 (__bf16*)d_ws);
        seq2seq_persistent_kernel<true><<<dim3(1), dim3(512), 0, stream>>>(p);
    } else {
        seq2seq_persistent_kernel<false><<<dim3(1), dim3(512), 0, stream>>>(p);
    }
}
